// CoupledMambaBlock_68401649156689
// MI455X (gfx1250) — matrix operand order repaired and checked by reading, not compiled, not run on hardware
//
#include <hip/hip_runtime.h>
#include <hip/hip_bf16.h>

// ---------------------------------------------------------------------------
// CoupledMambaBlock for MI455X (gfx1250, wave32, WMMA).
//  - All GEMMs: v_wmma_f32_16x16x32_bf16, f32 accumulate.
//  - Weights pre-packed (bf16, transposed, 16-padded) so each lane's B
//    fragment is two contiguous 16B loads; activations produced in
//    bf16 so each lane's A fragment is two contiguous 16B loads.
//  - No divergent loads anywhere near the WMMA (EXEC stays all-ones).
//  - SSM scan: 1 block per (batch, head, dir); thread p owns state row in
//    VGPRs; next-step row prefetched via global_prefetch.
// ---------------------------------------------------------------------------

#define SEQ     2304          // 48*48
#define BATCH   8
#define R_ROWS  (BATCH*SEQ)   // 18432 token rows
#define DM      192           // D_MODEL
#define DI      384           // D_INNER
#define DS      64            // D_STATE
#define NH      6             // NHEADS
#define HD      64            // HEADDIM
#define CD      512           // CONV_DIM
#define DPJ     902           // D_IN_PROJ (logical)
#define DPJP    912           // D_IN_PROJ padded to /16
#define DTOFF   896           // DPJ - NH

typedef __attribute__((ext_vector_type(16))) __bf16 v16bf;
typedef __attribute__((ext_vector_type(8)))  __bf16 v8bf;
typedef __attribute__((ext_vector_type(8)))  float  v8f;

// ---------------- wave32 helpers ----------------
__device__ __forceinline__ float wave_sum(float v) {
#pragma unroll
  for (int off = 16; off > 0; off >>= 1) v += __shfl_down(v, off, 32);
  return __shfl(v, 0, 32);
}
__device__ __forceinline__ float sigmoidf(float x) { return 1.f / (1.f + __expf(-x)); }
__device__ __forceinline__ float siluf(float x)    { return x * sigmoidf(x); }

// ---------------------------------------------------------------------------
// 16x16 tile of A(MxK bf16, row-major) * Wt(K x Npad weight, stored bf16
// TRANSPOSED: Wt[n*K + k]).  K multiple of 32.
// ISA 7.12.2 layouts -> per-lane fragments are K-contiguous:
//   A: lane m=l%16, half=l/16:  a[0..7]=A[m][kk+8h .. +7], a[8..15]=A[m][kk+16+8h .. +7]
//   B: lane n=l%16, half=l/16:  b[0..7]=Wt[n][kk+8h .. +7], b[8..15]=Wt[n][kk+16+8h .. +7]
// ---------------------------------------------------------------------------
__device__ __forceinline__ v8f wmma_tile(const __bf16* __restrict__ A,
                                         const __bf16* __restrict__ Wt,
                                         int K, int mBase, int nBase) {
  const int lane = threadIdx.x & 31;
  const int half = lane >> 4;
  const int l16  = lane & 15;
  const size_t arow = (size_t)(mBase + l16) * K;
  const size_t brow = (size_t)(nBase + l16) * K;
  v8f acc = {};
  for (int kk = 0; kk < K; kk += 32) {
    const v8bf a0 = *(const v8bf*)(A + arow + kk + 8 * half);
    const v8bf a1 = *(const v8bf*)(A + arow + kk + 16 + 8 * half);
    const v16bf a = __builtin_shufflevector(a0, a1, 0, 1, 2, 3, 4, 5, 6, 7,
                                            8, 9, 10, 11, 12, 13, 14, 15);
    const v8bf b0 = *(const v8bf*)(Wt + brow + kk + 8 * half);
    const v8bf b1 = *(const v8bf*)(Wt + brow + kk + 16 + 8 * half);
    const v16bf b = __builtin_shufflevector(b0, b1, 0, 1, 2, 3, 4, 5, 6, 7,
                                            8, 9, 10, 11, 12, 13, 14, 15);
    acc = __builtin_amdgcn_wmma_f32_16x16x32_bf16(false, a, false, b,
                                                  (short)0, acc, false, false);
  }
  return acc;
}

// mBase for 4-wave blocks tiling M; nBase from grid y.
__device__ __forceinline__ int tile_m() {
  return (blockIdx.x * 4 + (threadIdx.x >> 5)) * 16;
}

// ---------------------------------------------------------------------------
// Pack f32 weight W (K x N, row-major) -> bf16 transposed padded Wt (Npad x K).
// ---------------------------------------------------------------------------
__global__ void k_pack_wt(const float* __restrict__ W, __bf16* __restrict__ Wt,
                          int K, int N, int Npad) {
  const int idx = blockIdx.x * blockDim.x + threadIdx.x;
  if (idx >= Npad * K) return;
  const int n = idx / K, k = idx % K;
  Wt[(size_t)n * K + k] = (n < N) ? (__bf16)W[(size_t)k * N + n] : (__bf16)0.f;
}

// ---------------------------------------------------------------------------
// LayerNorm over DM=192; one wave per row. Writes f32 (epilogue use) + bf16
// (GEMM-A use). img_layout=1: input is (B, DM, SEQ); else row-major.
// ---------------------------------------------------------------------------
__global__ void k_ln(const float* __restrict__ x, const float* __restrict__ g,
                     const float* __restrict__ bb, float* __restrict__ out32,
                     __bf16* __restrict__ out16, int img_layout) {
  const int row  = blockIdx.x;
  const int b    = row / SEQ, l = row % SEQ;
  const int lane = threadIdx.x;
  float v[6];
  float s = 0.f;
#pragma unroll
  for (int i = 0; i < 6; ++i) {
    const int d = lane + 32 * i;
    v[i] = img_layout ? x[((size_t)b * DM + d) * SEQ + l]
                      : x[(size_t)row * DM + d];
    s += v[i];
  }
  const float mean = wave_sum(s) * (1.f / DM);
  float vs = 0.f;
#pragma unroll
  for (int i = 0; i < 6; ++i) { const float t = v[i] - mean; vs += t * t; }
  const float inv = rsqrtf(wave_sum(vs) * (1.f / DM) + 1e-5f);
#pragma unroll
  for (int i = 0; i < 6; ++i) {
    const int d = lane + 32 * i;
    const float o = (v[i] - mean) * inv * g[d] + bb[d];
    out32[(size_t)row * DM + d] = o;
    out16[(size_t)row * DM + d] = (__bf16)o;
  }
}

// ---------------------------------------------------------------------------
// Xin(bf16) = clip(Xn + scale * sigmoid(Yn @ W + bias), -2, 2).  K=N=192.
// ---------------------------------------------------------------------------
__global__ void k_cross(const __bf16* __restrict__ Yn, const float* __restrict__ Xn,
                        const __bf16* __restrict__ Wt, const float* __restrict__ bias,
                        const float* __restrict__ scale, __bf16* __restrict__ Xin) {
  const int mBase = tile_m(), nBase = blockIdx.y * 16;
  v8f acc = wmma_tile(Yn, Wt, DM, mBase, nBase);
  const int lane = threadIdx.x & 31, half = lane >> 4, n = nBase + (lane & 15);
  const float a = *scale;
#pragma unroll
  for (int r = 0; r < 8; ++r) {
    const int m = mBase + r + 8 * half;
    float val = Xn[(size_t)m * DM + n] + a * sigmoidf(acc[r] + bias[n]);
    val = fminf(2.f, fmaxf(-2.f, val));
    Xin[(size_t)m * DM + n] = (__bf16)val;
  }
}

// ---------------------------------------------------------------------------
// In-projection: zx(f32, stride DPJP) = Xin @ W_in. K=192, N tiles cover 912.
// ---------------------------------------------------------------------------
__global__ void k_inproj(const __bf16* __restrict__ A, const __bf16* __restrict__ Wt,
                         float* __restrict__ C) {
  const int mBase = tile_m(), nBase = blockIdx.y * 16;
  v8f acc = wmma_tile(A, Wt, DM, mBase, nBase);
  const int lane = threadIdx.x & 31, half = lane >> 4, n = nBase + (lane & 15);
#pragma unroll
  for (int r = 0; r < 8; ++r) {
    const int m = mBase + r + 8 * half;
    C[(size_t)m * DPJP + n] = acc[r];
  }
}

// ---------------------------------------------------------------------------
// Depthwise causal conv (width 4) over zx[:, 384:896] + SiLU.
// ---------------------------------------------------------------------------
__global__ void k_conv_silu(const float* __restrict__ zx, const float* __restrict__ cw,
                            const float* __restrict__ cb, float* __restrict__ xbc) {
  const int idx = blockIdx.x * blockDim.x + threadIdx.x;
  if (idx >= R_ROWS * CD) return;
  const int c = idx % CD, row = idx / CD;
  const int b = row / SEQ, l = row % SEQ;
  float acc = cb[c];
#pragma unroll
  for (int k = 0; k < 4; ++k) {
    const int ls = l - 3 + k;
    if (ls >= 0)
      acc += zx[(size_t)(b * SEQ + ls) * DPJP + (DI + c)] * cw[k * CD + c];
  }
  xbc[(size_t)row * CD + c] = siluf(acc);
}

// ---------------------------------------------------------------------------
// Sequential SSM scan. One block per (batch, head); 64 threads; thread p owns
// state row h[p][0..63] in registers so y[p] is a thread-local reduction.
// dir=0 forward, dir=1 backward (output already un-flipped).
// ---------------------------------------------------------------------------
__global__ void k_scan(const float* __restrict__ xbc, const float* __restrict__ zx,
                       const float* __restrict__ dt_bias, const float* __restrict__ A_log,
                       float* __restrict__ yout, int dir) {
  const int blk = blockIdx.x;
  const int b = blk / NH, hh = blk % NH;
  const int p = threadIdx.x;                 // 0..63
  const float Ah = -__expf(A_log[hh]);
  float h[DS];
#pragma unroll
  for (int s = 0; s < DS; ++s) h[s] = 0.f;
  __shared__ float Bs[DS], Cs[DS], sc[2];
  const int step = dir ? -1 : 1;
  int l = dir ? (SEQ - 1) : 0;
  for (int t = 0; t < SEQ; ++t, l += step) {
    const size_t rowc = (size_t)(b * SEQ + l) * CD;
    // Prefetch next timestep's row while this step's serial chain executes.
    if (t + 1 < SEQ)
      __builtin_prefetch(xbc + rowc + (size_t)step * CD + DI + p, 0, 1);
    Bs[p] = xbc[rowc + DI + p];
    Cs[p] = xbc[rowc + DI + DS + p];
    if (p == 0) {
      const float dtr = zx[(size_t)(b * SEQ + l) * DPJP + DTOFF + hh] + dt_bias[hh];
      const float dtv = (dtr > 0.f) ? (dtr + log1pf(__expf(-dtr)))
                                    : log1pf(__expf(dtr));
      sc[0] = dtv;
      sc[1] = __expf(dtv * Ah);
    }
    __syncthreads();
    const float dtv = sc[0], dA = sc[1];
    const float coef = dtv * xbc[rowc + hh * HD + p];
    float acc = 0.f;
#pragma unroll
    for (int s = 0; s < DS; ++s) {
      h[s] = h[s] * dA + coef * Bs[s];
      acc += h[s] * Cs[s];
    }
    yout[((size_t)(b * SEQ + l) * NH + hh) * HD + p] = acc;
    __syncthreads();
  }
}

// ---------------------------------------------------------------------------
// y = yf + yb + D*xh ; u = y*silu(z) ; RMSNorm(384)*norm_w -> un (bf16).
// ---------------------------------------------------------------------------
__global__ void k_gate_rms(const float* __restrict__ yf, const float* __restrict__ yb,
                           const float* __restrict__ xbc, const float* __restrict__ zx,
                           const float* __restrict__ Dp,  const float* __restrict__ nw,
                           __bf16* __restrict__ un) {
  const int row = blockIdx.x, lane = threadIdx.x;
  const size_t base = (size_t)row * DI;
  float u[12];
  float ss = 0.f;
#pragma unroll
  for (int i = 0; i < 12; ++i) {
    const int c = lane + 32 * i;
    const float xh = xbc[(size_t)row * CD + c];
    const float y  = yf[base + c] + yb[base + c] + Dp[c / HD] * xh;
    const float z  = zx[(size_t)row * DPJP + c];
    u[i] = y * siluf(z);
    ss += u[i] * u[i];
  }
  const float inv = rsqrtf(wave_sum(ss) * (1.f / DI) + 1e-5f);
#pragma unroll
  for (int i = 0; i < 12; ++i) {
    const int c = lane + 32 * i;
    un[base + c] = (__bf16)(u[i] * inv * nw[c]);
  }
}

// ---------------------------------------------------------------------------
// out-proj + residual: xr(f32) = img_flat + un @ Wout. K=384, N=192.
// ---------------------------------------------------------------------------
__global__ void k_outproj(const __bf16* __restrict__ un, const __bf16* __restrict__ Wt,
                          const float* __restrict__ img, float* __restrict__ xr) {
  const int mBase = tile_m(), nBase = blockIdx.y * 16;
  v8f acc = wmma_tile(un, Wt, DI, mBase, nBase);
  const int lane = threadIdx.x & 31, half = lane >> 4, n = nBase + (lane & 15);
#pragma unroll
  for (int r = 0; r < 8; ++r) {
    const int m = mBase + r + 8 * half;
    const int b = m / SEQ, l = m % SEQ;
    xr[(size_t)m * DM + n] = img[((size_t)b * DM + n) * SEQ + l] + acc[r];
  }
}

// ---------------------------------------------------------------------------
// FFN layer 1: fh(bf16) = gelu_exact(A @ W1 + b1). K=192, N=384.
// ---------------------------------------------------------------------------
__global__ void k_gemm_gelu(const __bf16* __restrict__ A, const __bf16* __restrict__ Wt,
                            const float* __restrict__ bias, __bf16* __restrict__ C) {
  const int mBase = tile_m(), nBase = blockIdx.y * 16;
  v8f acc = wmma_tile(A, Wt, DM, mBase, nBase);
  const int lane = threadIdx.x & 31, half = lane >> 4, n = nBase + (lane & 15);
#pragma unroll
  for (int r = 0; r < 8; ++r) {
    const int m = mBase + r + 8 * half;
    const float x = acc[r] + bias[n];
    C[(size_t)m * DI + n] = (__bf16)(0.5f * x * (1.f + erff(x * 0.70710678118654752f)));
  }
}

// ---------------------------------------------------------------------------
// FFN layer 2 + residual, scattered to image-layout output. K=384, N=192.
// ---------------------------------------------------------------------------
__global__ void k_ffn2(const __bf16* __restrict__ A, const __bf16* __restrict__ Wt,
                       const float* __restrict__ bias, const float* __restrict__ xr,
                       float* __restrict__ out) {
  const int mBase = tile_m(), nBase = blockIdx.y * 16;
  v8f acc = wmma_tile(A, Wt, DI, mBase, nBase);
  const int lane = threadIdx.x & 31, half = lane >> 4, n = nBase + (lane & 15);
#pragma unroll
  for (int r = 0; r < 8; ++r) {
    const int m = mBase + r + 8 * half;
    const int b = m / SEQ, l = m % SEQ;
    out[((size_t)b * DM + n) * SEQ + l] = xr[(size_t)m * DM + n] + acc[r] + bias[n];
  }
}

// ---------------------------------------------------------------------------
// Host-side orchestration. Input order (setup_inputs insertion order):
//  0 sem 1 depth | 2..5 ln g/b | 6 cd2s_w 7 cd2s_b 8 cs2d_w 9 cs2d_b
//  10 alpha 11 beta | 12..19 sem_mamba {W_in,conv_w,conv_b,dt_bias,A_log,D,
//  norm_w,W_out} | 20..27 depth_mamba | 28..33 ffn_sem {g,b,w1,b1,w2,b2} |
//  34..39 ffn_depth
// ---------------------------------------------------------------------------
extern "C" void kernel_launch(void* const* d_in, const int* in_sizes, int n_in,
                              void* d_out, int out_size, void* d_ws, size_t ws_size,
                              hipStream_t stream) {
  (void)in_sizes; (void)n_in; (void)out_size; (void)ws_size;
  auto f = [&](int i) { return (const float*)d_in[i]; };
  const size_t R = R_ROWS;

  char* base = (char*)d_ws;
  size_t off = 0;
  auto carve = [&](size_t bytes) -> char* {
    char* r = base + off;
    off = (off + bytes + 255) & ~(size_t)255;
    return r;
  };
  // f32 activations
  float* xn0_32 = (float*)carve(R * DM * 4);   // sem_n (epilogue) / FFN LN scratch
  float* xn1_32 = (float*)carve(R * DM * 4);   // depth_n
  float* zx     = (float*)carve(R * DPJP * 4); // zxbcdt (padded stride)
  float* xbc    = (float*)carve(R * CD * 4);   // conv+silu output
  float* yf     = (float*)carve(R * DI * 4);
  float* yb     = (float*)carve(R * DI * 4);
  float* xr0    = (float*)carve(R * DM * 4);
  float* xr1    = (float*)carve(R * DM * 4);
  // bf16 GEMM-A activations
  __bf16* xn0_16 = (__bf16*)carve(R * DM * 2);
  __bf16* xn1_16 = (__bf16*)carve(R * DM * 2);
  __bf16* xin0   = (__bf16*)carve(R * DM * 2);
  __bf16* xin1   = (__bf16*)carve(R * DM * 2);
  __bf16* un16   = (__bf16*)carve(R * DI * 2);
  __bf16* lnb16  = (__bf16*)carve(R * DM * 2);
  __bf16* fh16   = (__bf16*)carve(R * DI * 2);
  // bf16 transposed packed weights
  __bf16* cd2s_t = (__bf16*)carve((size_t)DM * DM * 2);
  __bf16* cs2d_t = (__bf16*)carve((size_t)DM * DM * 2);
  __bf16* win_t  = (__bf16*)carve((size_t)DPJP * DM * 2);
  __bf16* wout_t = (__bf16*)carve((size_t)DM * DI * 2);
  __bf16* w1_t   = (__bf16*)carve((size_t)DI * DM * 2);
  __bf16* w2_t   = (__bf16*)carve((size_t)DM * DI * 2);

  const dim3 wv(32);
  const dim3 gw(128);                       // 4 waves / block along M
  auto packGrid = [](int K, int Npad) { return dim3((Npad * K + 255) / 256); };

  // ---- pack cross-gate weights ----
  k_pack_wt<<<packGrid(DM, DM), 256, 0, stream>>>(f(6), cd2s_t, DM, DM, DM);
  k_pack_wt<<<packGrid(DM, DM), 256, 0, stream>>>(f(8), cs2d_t, DM, DM, DM);

  // ---- LayerNorm both streams ----
  k_ln<<<R, wv, 0, stream>>>(f(0), f(2), f(3), xn0_32, xn0_16, 1);
  k_ln<<<R, wv, 0, stream>>>(f(1), f(4), f(5), xn1_32, xn1_16, 1);

  // ---- cross gates ----
  const dim3 gCross(R / 64, DM / 16);
  k_cross<<<gCross, gw, 0, stream>>>(xn1_16, xn0_32, cd2s_t, f(7), f(10), xin0);
  k_cross<<<gCross, gw, 0, stream>>>(xn0_16, xn1_32, cs2d_t, f(9), f(11), xin1);

  // ---- Mamba-2 per stream (big scratch reused sequentially) ----
  for (int s = 0; s < 2; ++s) {
    const int pb = 12 + 8 * s;
    const __bf16* Ain = s ? xin1 : xin0;
    const float* img = f(s);
    float* xr = s ? xr1 : xr0;

    k_pack_wt<<<packGrid(DM, DPJP), 256, 0, stream>>>(f(pb + 0), win_t, DM, DPJ, DPJP);
    k_pack_wt<<<packGrid(DI, DM),   256, 0, stream>>>(f(pb + 7), wout_t, DI, DM, DM);

    k_inproj<<<dim3(R / 64, DPJP / 16), gw, 0, stream>>>(Ain, win_t, zx);
    const int tot = R_ROWS * CD;
    k_conv_silu<<<(tot + 255) / 256, 256, 0, stream>>>(zx, f(pb + 1), f(pb + 2), xbc);
    k_scan<<<BATCH * NH, HD, 0, stream>>>(xbc, zx, f(pb + 3), f(pb + 4), yf, 0);
    k_scan<<<BATCH * NH, HD, 0, stream>>>(xbc, zx, f(pb + 3), f(pb + 4), yb, 1);
    k_gate_rms<<<R, wv, 0, stream>>>(yf, yb, xbc, zx, f(pb + 5), f(pb + 6), un16);
    k_outproj<<<dim3(R / 64, DM / 16), gw, 0, stream>>>(un16, wout_t, img, xr);
  }

  // ---- FFN + final residual per stream, scatter to image-layout output ----
  for (int s = 0; s < 2; ++s) {
    const int fb = 28 + 6 * s;
    float* xr = s ? xr1 : xr0;
    float* outp = (float*)d_out + (size_t)s * R * DM;

    k_pack_wt<<<packGrid(DM, DI), 256, 0, stream>>>(f(fb + 2), w1_t, DM, DI, DI);
    k_pack_wt<<<packGrid(DI, DM), 256, 0, stream>>>(f(fb + 4), w2_t, DI, DM, DM);

    k_ln<<<R, wv, 0, stream>>>(xr, f(fb + 0), f(fb + 1), xn0_32 /*scratch*/, lnb16, 0);
    k_gemm_gelu<<<dim3(R / 64, DI / 16), gw, 0, stream>>>(lnb16, w1_t, f(fb + 3), fh16);
    k_ffn2<<<dim3(R / 64, DM / 16), gw, 0, stream>>>(fh16, w2_t, f(fb + 5), xr, outp);
  }
}
